// EEG_VQ_VAE_33079838114251
// MI455X (gfx1250) — matrix operand order repaired and checked by reading, compiled, not run on hardware
//
#include <hip/hip_runtime.h>
#include <hip/hip_bf16.h>
#include <math.h>

typedef __bf16 bf16_t;
typedef __attribute__((ext_vector_type(16))) __bf16        v16bf;
typedef __attribute__((ext_vector_type(8)))  float         v8f;
typedef __attribute__((ext_vector_type(4)))  unsigned int  u32x4;
typedef __attribute__((ext_vector_type(4)))  int           v4i;

constexpr int Bq = 65536;   // batch
constexpr int Dq = 128;     // input dim
constexpr int Hq = 512;     // hidden
constexpr int Eq = 64;      // embedding dim
constexpr int Kq = 4096;    // codebook size

// ---------------------------------------------------------------------------
// Async global->LDS copy path (CDNA5). Guarded: falls back to load+ds_store.
// Builtin prototype (from hipcc diagnostic): (v4i AS1* src, v4i AS3* dst,
// imm offset, imm cpol).
// ---------------------------------------------------------------------------
#if defined(__has_builtin)
#  if __has_builtin(__builtin_amdgcn_global_load_async_to_lds_b128)
#    define USE_ASYNC_LDS 1
#  endif
#endif
#ifndef USE_ASYNC_LDS
#  define USE_ASYNC_LDS 0
#endif

#if USE_ASYNC_LDS
#define AS_GLOBAL __attribute__((address_space(1)))
#define AS_LDS    __attribute__((address_space(3)))
__device__ __forceinline__ void async_copy16(const void* g, void* l) {
  __builtin_amdgcn_global_load_async_to_lds_b128(
      (AS_GLOBAL v4i*)g, (AS_LDS v4i*)l, 0, 0);
}
#  if __has_builtin(__builtin_amdgcn_s_wait_asynccnt)
__device__ __forceinline__ void async_wait0() {
  __builtin_amdgcn_s_wait_asynccnt(0);
}
#  else
__device__ __forceinline__ void async_wait0() {
  asm volatile("s_wait_asynccnt 0x0" ::: "memory");
}
#  endif
#endif

// ---------------------------------------------------------------------------
// fp32 -> bf16 elementwise convert
// ---------------------------------------------------------------------------
__global__ __launch_bounds__(256) void cvt_f32_bf16(const float* __restrict__ in,
                                                    bf16_t* __restrict__ out, int n) {
  int i = blockIdx.x * 256 + threadIdx.x;
  if (i < n) out[i] = (bf16_t)in[i];
}

// ---------------------------------------------------------------------------
// fp32 [Kd][N] -> bf16 transposed [N][Kd] (one-time weight prep; makes the
// GEMM B-tile a contiguous-copy, async-LDS-friendly layout)
// ---------------------------------------------------------------------------
__global__ __launch_bounds__(256) void cvt_transpose_bf16(const float* __restrict__ in,
                                                          bf16_t* __restrict__ out,
                                                          int Kd, int N) {
  int t = blockIdx.x * 256 + threadIdx.x;
  if (t < Kd * N) {
    int k = t / N, n = t - k * N;
    out[(size_t)n * Kd + k] = (bf16_t)in[t];
  }
}

// ---------------------------------------------------------------------------
// Fold BatchNorm (eval) into per-column scale/shift for layer 1 epilogue
// ---------------------------------------------------------------------------
__global__ __launch_bounds__(256) void bn_fold(const float* __restrict__ b1,
                                               const float* __restrict__ g1,
                                               const float* __restrict__ be1,
                                               const float* __restrict__ rm1,
                                               const float* __restrict__ rv1,
                                               float* __restrict__ scale,
                                               float* __restrict__ shift, int n) {
  int i = blockIdx.x * 256 + threadIdx.x;
  if (i < n) {
    float sc = g1[i] * rsqrtf(rv1[i] + 1e-5f);
    scale[i] = sc;
    shift[i] = (b1[i] - rm1[i]) * sc + be1[i];
  }
}

// ---------------------------------------------------------------------------
// Row L2-normalize (rows of length 64), fp32 in -> bf16 out. One wave per row.
// ---------------------------------------------------------------------------
__global__ __launch_bounds__(256) void rownorm64_bf16(const float* __restrict__ in,
                                                      bf16_t* __restrict__ out, int rows) {
  const int wave = threadIdx.x >> 5;
  const int lane = threadIdx.x & 31;
  const int row  = blockIdx.x * 8 + wave;
  if (row >= rows) return;
  const float* p = in + (size_t)row * 64;
  float v0 = p[lane], v1 = p[lane + 32];
  float s = v0 * v0 + v1 * v1;
  #pragma unroll
  for (int off = 16; off >= 1; off >>= 1) s += __shfl_xor(s, off, 32);
  float r = 1.0f / fmaxf(sqrtf(s), 1e-12f);
  out[(size_t)row * 64 + lane]      = (bf16_t)(v0 * r);
  out[(size_t)row * 64 + lane + 32] = (bf16_t)(v1 * r);
}

// ---------------------------------------------------------------------------
// Zero histogram + loss accumulator (must run every launch: graph replays)
// ---------------------------------------------------------------------------
__global__ __launch_bounds__(256) void vq_zero(unsigned int* __restrict__ hist,
                                               float* __restrict__ lossAcc) {
  int t = blockIdx.x * 256 + threadIdx.x;
  if (t < Kq) hist[t] = 0u;
  if (t == Kq) lossAcc[0] = 0.0f;
}

// ---------------------------------------------------------------------------
// BF16 WMMA GEMM: Out[M,N] = epilogue(A[M,Kd] @ W[Kd,N]), W given TRANSPOSED
// as Wt[N][Kd] (k-contiguous).
//   MODE 0: + bias, ReLU, bf16 out
//   MODE 1: * scale + shift (folded BN), ReLU, bf16 out
//   MODE 2: + bias, fp32 out (no ReLU)
// Block tile 128x64, 8 waves, each wave 32x32 (2x2 WMMA frags), BK=32.
// Double-buffered LDS; staging via async global->LDS b128 when available.
// ---------------------------------------------------------------------------
template <int MODE>
__global__ __launch_bounds__(256) void gemm_bf16_wmma(
    const bf16_t* __restrict__ A, const bf16_t* __restrict__ Wt,
    const float* __restrict__ bias, const float* __restrict__ scale,
    void* __restrict__ Out, int M, int Kd, int N) {
  __shared__ bf16_t As[2][128 * 40];   // row-major, padded stride 40
  __shared__ bf16_t Bs[2][64 * 40];    // [n][k], k-contiguous, padded stride 40

  const int tid  = threadIdx.x;
  const int wave = tid >> 5;
  const int lane = tid & 31;
  const int half = lane >> 4;
  const int l16  = lane & 15;
  const int wm = (wave & 3) * 32;
  const int wn = (wave >> 2) * 32;
  const int m0 = blockIdx.x * 128;
  const int n0 = blockIdx.y * 64;

  // stage one BK=32 slab (A: 128x32, B: 64x32) into LDS buffer `buf`
  auto stage = [&](int buf, int k0) {
    #pragma unroll
    for (int i = 0; i < 2; ++i) {           // A: 512 x 16B chunks, 2/thread
      int idx = tid + i * 256;
      int r = idx >> 2;
      int c = (idx & 3) * 8;
      const bf16_t* src = A + (size_t)(m0 + r) * Kd + (k0 + c);
      bf16_t* dst = &As[buf][r * 40 + c];
#if USE_ASYNC_LDS
      async_copy16(src, dst);
#else
      *(u32x4*)dst = *(const u32x4*)src;
#endif
    }
    {                                        // B: 256 x 16B chunks, 1/thread
      int nn = tid >> 2;
      int cc = (tid & 3) * 8;
      const bf16_t* src = Wt + (size_t)(n0 + nn) * Kd + (k0 + cc);
      bf16_t* dst = &Bs[buf][nn * 40 + cc];
#if USE_ASYNC_LDS
      async_copy16(src, dst);
#else
      *(u32x4*)dst = *(const u32x4*)src;
#endif
    }
  };

  v8f acc[2][2] = {};
  const int nk = Kd >> 5;
  stage(0, 0);

  for (int it = 0; it < nk; ++it) {
    const int cur = it & 1;
#if USE_ASYNC_LDS
    async_wait0();                 // our copies for tile `it` have landed
#endif
    __syncthreads();               // everyone's copies landed / prior reads done
    if (it + 1 < nk) stage(cur ^ 1, (it + 1) << 5);   // overlap next copy

    // --- build fragments, contiguous k = 16*half + s in both A and B ---
    v16bf afrag[2], bfrag[2];
    #pragma unroll
    for (int t = 0; t < 2; ++t) {   // A 16x32: lane holds row (l16), K per half
      const bf16_t* p = &As[cur][(wm + t * 16 + l16) * 40];
      union { v16bf v; u32x4 q[2]; } u;
      u.q[0] = *(const u32x4*)(p + 16 * half);       // K = 16*half .. +7
      u.q[1] = *(const u32x4*)(p + 16 * half + 8);   // K = 16*half+8 .. +7
      afrag[t] = u.v;
    }
    #pragma unroll
    for (int t = 0; t < 2; ++t) {   // B 32x16: lane holds col (l16), K = 16*half..+15
      const bf16_t* p = &Bs[cur][(wn + t * 16 + l16) * 40 + 16 * half];
      union { v16bf v; u32x4 q[2]; } u;
      u.q[0] = *(const u32x4*)p;
      u.q[1] = *(const u32x4*)(p + 8);
      bfrag[t] = u.v;
    }
    #pragma unroll
    for (int i = 0; i < 2; ++i)
      #pragma unroll
      for (int j = 0; j < 2; ++j)
        acc[i][j] = __builtin_amdgcn_wmma_f32_16x16x32_bf16(
            false, afrag[i], false, bfrag[j], (short)0, acc[i][j], false, false);
  }

  // --- epilogue: C 16x16 layout: VGPR r -> row r+8*half, lane -> col l16 ---
  #pragma unroll
  for (int i = 0; i < 2; ++i) {
    #pragma unroll
    for (int j = 0; j < 2; ++j) {
      #pragma unroll
      for (int r = 0; r < 8; ++r) {
        int row = m0 + wm + i * 16 + r + 8 * half;
        int col = n0 + wn + j * 16 + l16;
        float v = acc[i][j][r];
        if (MODE == 1) v = v * scale[col] + bias[col];
        else           v = v + bias[col];
        if (MODE != 2) v = fmaxf(v, 0.0f);
        if (MODE == 2) ((float*)Out)[(size_t)row * N + col] = v;
        else           ((bf16_t*)Out)[(size_t)row * N + col] = (bf16_t)v;
      }
    }
  }
}

// ---------------------------------------------------------------------------
// VQ scoring via WMMA: argmax_k <zn[m], en[k]>  (== argmin cosine distance).
// Each wave owns 16 rows (2 A frags, K=64); streams all 4096 codes as B frags
// directly from global (en rows are the contraction dim -> column-major B).
// ---------------------------------------------------------------------------
__global__ __launch_bounds__(256) void vq_score_wmma(
    const bf16_t* __restrict__ zn, const bf16_t* __restrict__ en,
    int* __restrict__ idx) {
  const int tid  = threadIdx.x;
  const int wave = tid >> 5;
  const int lane = tid & 31;
  const int half = lane >> 4;
  const int l16  = lane & 15;
  const int rowbase = blockIdx.x * 128 + wave * 16;

  v16bf a[2];
  #pragma unroll
  for (int t = 0; t < 2; ++t) {
    const bf16_t* p = zn + (size_t)(rowbase + l16) * 64 + t * 32;
    union { v16bf v; u32x4 q[2]; } u;
    u.q[0] = *(const u32x4*)(p + 16 * half);
    u.q[1] = *(const u32x4*)(p + 16 * half + 8);
    a[t] = u.v;
  }

  float best[8];
  int   bidx[8];
  #pragma unroll
  for (int r = 0; r < 8; ++r) { best[r] = -3.0e38f; bidx[r] = 0; }

  for (int ncol = 0; ncol < Kq; ncol += 16) {
    v16bf b[2];
    const bf16_t* er = en + (size_t)(ncol + l16) * 64;
    #pragma unroll
    for (int t = 0; t < 2; ++t) {
      const bf16_t* p = er + t * 32 + 16 * half;
      union { v16bf v; u32x4 q[2]; } u;
      u.q[0] = *(const u32x4*)p;
      u.q[1] = *(const u32x4*)(p + 8);
      b[t] = u.v;
    }
    v8f c = {};
    c = __builtin_amdgcn_wmma_f32_16x16x32_bf16(false, a[0], false, b[0], (short)0, c, false, false);
    c = __builtin_amdgcn_wmma_f32_16x16x32_bf16(false, a[1], false, b[1], (short)0, c, false, false);
    int mycol = ncol + l16;
    #pragma unroll
    for (int r = 0; r < 8; ++r) {
      if (c[r] > best[r]) { best[r] = c[r]; bidx[r] = mycol; }
    }
  }

  // cross-lane argmax within each 16-lane half (ties -> lowest index)
  #pragma unroll
  for (int off = 8; off >= 1; off >>= 1) {
    #pragma unroll
    for (int r = 0; r < 8; ++r) {
      float ov = __shfl_xor(best[r], off, 32);
      int   oi = __shfl_xor(bidx[r], off, 32);
      if (ov > best[r] || (ov == best[r] && oi < bidx[r])) { best[r] = ov; bidx[r] = oi; }
    }
  }
  if (l16 == 0) {
    #pragma unroll
    for (int r = 0; r < 8; ++r) idx[rowbase + r + 8 * half] = bidx[r];
  }
}

// ---------------------------------------------------------------------------
// Gather q = emb[idx], emit bf16 q for decoder, accumulate sum((q-z)^2) and
// the code histogram. One thread per (row, e) element.
// ---------------------------------------------------------------------------
__global__ __launch_bounds__(256) void vq_gather(
    const int* __restrict__ idx, const float* __restrict__ emb,
    const float* __restrict__ z, bf16_t* __restrict__ qb,
    float* __restrict__ lossAcc, unsigned int* __restrict__ hist) {
  size_t t = (size_t)blockIdx.x * 256 + threadIdx.x;
  int row = (int)(t >> 6);
  int e   = (int)(t & 63);
  int ci  = idx[row];
  float q = emb[(size_t)ci * 64 + e];
  float d = q - z[t];
  qb[t] = (bf16_t)q;
  float s = d * d;
  #pragma unroll
  for (int off = 16; off >= 1; off >>= 1) s += __shfl_xor(s, off, 32);
  if ((threadIdx.x & 31) == 0) atomicAdd(lossAcc, s);
  if (e == 0) atomicAdd(&hist[ci], 1u);
}

// ---------------------------------------------------------------------------
// Finalize scalars: vq_loss = 1.02 * mean((q-z)^2); perplexity from histogram
// ---------------------------------------------------------------------------
__global__ __launch_bounds__(256) void vq_finalize(
    const float* __restrict__ lossAcc, const unsigned int* __restrict__ hist,
    float* __restrict__ out_loss, float* __restrict__ out_perp) {
  __shared__ float red[256];
  float s = 0.0f;
  for (int k = threadIdx.x; k < Kq; k += 256) {
    float p = (float)hist[k] * (1.0f / (float)Bq);
    s += p * logf(p + 1e-10f);
  }
  red[threadIdx.x] = s;
  __syncthreads();
  for (int st = 128; st > 0; st >>= 1) {
    if (threadIdx.x < st) red[threadIdx.x] += red[threadIdx.x + st];
    __syncthreads();
  }
  if (threadIdx.x == 0) {
    *out_perp = expf(-red[0]);
    *out_loss = (1.0f + 0.02f) * lossAcc[0] * (1.0f / ((float)Bq * (float)Eq));
  }
}

// ---------------------------------------------------------------------------
extern "C" void kernel_launch(void* const* d_in, const int* in_sizes, int n_in,
                              void* d_out, int out_size, void* d_ws, size_t ws_size,
                              hipStream_t stream) {
  (void)in_sizes; (void)n_in; (void)out_size; (void)ws_size;

  const float* x   = (const float*)d_in[0];
  const float* W1  = (const float*)d_in[1];
  const float* b1  = (const float*)d_in[2];
  const float* g1  = (const float*)d_in[3];
  const float* be1 = (const float*)d_in[4];
  const float* rm1 = (const float*)d_in[5];
  const float* rv1 = (const float*)d_in[6];
  const float* W2  = (const float*)d_in[7];
  const float* b2  = (const float*)d_in[8];
  const float* W3  = (const float*)d_in[9];
  const float* b3  = (const float*)d_in[10];
  const float* W4  = (const float*)d_in[11];
  const float* b4  = (const float*)d_in[12];
  const float* emb = (const float*)d_in[13];
  const float* D1  = (const float*)d_in[14];
  const float* db1 = (const float*)d_in[15];
  const float* D2  = (const float*)d_in[16];
  const float* db2 = (const float*)d_in[17];
  const float* D3  = (const float*)d_in[18];
  const float* db3 = (const float*)d_in[19];
  const float* D4  = (const float*)d_in[20];
  const float* db4 = (const float*)d_in[21];

  char* ws = (char*)d_ws;
  size_t off = 0;
  auto take = [&](size_t bytes) -> char* {
    char* p = ws + off;
    off += (bytes + 255) & ~(size_t)255;
    return p;
  };

  bf16_t* xb   = (bf16_t*)take((size_t)Bq * Dq * 2);          // x bf16 (reused below)
  bf16_t* W1t  = (bf16_t*)take((size_t)Dq * Hq * 2);          // [512][128]
  bf16_t* W2t  = (bf16_t*)take((size_t)Hq * Hq * 2);          // [512][512]
  bf16_t* W3t  = (bf16_t*)take((size_t)Hq * (Hq / 2) * 2);    // [256][512]
  bf16_t* W4t  = (bf16_t*)take((size_t)(Hq / 2) * Eq * 2);    // [64][256]
  bf16_t* D1t  = (bf16_t*)take((size_t)Eq * (Hq / 2) * 2);    // [256][64]
  bf16_t* D2t  = (bf16_t*)take((size_t)(Hq / 2) * Hq * 2);    // [512][256]
  bf16_t* D3t  = (bf16_t*)take((size_t)Hq * Hq * 2);          // [512][512]
  bf16_t* D4t  = (bf16_t*)take((size_t)Hq * Dq * 2);          // [128][512]
  bf16_t* enb  = (bf16_t*)take((size_t)Kq * Eq * 2);
  float*  scale1 = (float*)take(Hq * 4);
  float*  shift1 = (float*)take(Hq * 4);
  bf16_t* actA = (bf16_t*)take((size_t)Bq * Hq * 2);
  bf16_t* actB = (bf16_t*)take((size_t)Bq * Hq * 2);
  float*  zf   = (float*)take((size_t)Bq * Eq * 4);
  int*    idxb = (int*)take((size_t)Bq * 4);
  unsigned int* hist = (unsigned int*)take(Kq * 4);
  float*  lossAcc = (float*)take(256);
  // x bf16 buffer is dead after layer 1 -> reuse for zn (bf16) and q (bf16)
  bf16_t* znb = xb;                       // B*64 bf16
  bf16_t* qb  = xb + (size_t)Bq * Eq;

  float* outp = (float*)d_out;            // [vq_loss, x_recon(B*D), perplexity]

  dim3 blk(256);
  auto cvtT = [&](const float* src, bf16_t* dst, int Kd, int N) {
    cvt_transpose_bf16<<<dim3((Kd * N + 255) / 256), blk, 0, stream>>>(src, dst, Kd, N);
  };

  // ---- prep ----
  cvt_f32_bf16<<<dim3((Bq * Dq + 255) / 256), blk, 0, stream>>>(x, xb, Bq * Dq);
  cvtT(W1, W1t, Dq, Hq);
  cvtT(W2, W2t, Hq, Hq);
  cvtT(W3, W3t, Hq, Hq / 2);
  cvtT(W4, W4t, Hq / 2, Eq);
  cvtT(D1, D1t, Eq, Hq / 2);
  cvtT(D2, D2t, Hq / 2, Hq);
  cvtT(D3, D3t, Hq, Hq);
  cvtT(D4, D4t, Hq, Dq);
  bn_fold<<<dim3((Hq + 255) / 256), blk, 0, stream>>>(b1, g1, be1, rm1, rv1, scale1, shift1, Hq);
  rownorm64_bf16<<<dim3(Kq / 8), blk, 0, stream>>>(emb, enb, Kq);
  vq_zero<<<dim3(Kq / 256 + 1), blk, 0, stream>>>(hist, lossAcc);

  // ---- encoder ----
  gemm_bf16_wmma<1><<<dim3(Bq / 128, Hq / 64), blk, 0, stream>>>(
      xb, W1t, shift1, scale1, actA, Bq, Dq, Hq);                       // BN+ReLU
  gemm_bf16_wmma<0><<<dim3(Bq / 128, Hq / 64), blk, 0, stream>>>(
      actA, W2t, b2, nullptr, actB, Bq, Hq, Hq);
  gemm_bf16_wmma<0><<<dim3(Bq / 128, (Hq / 2) / 64), blk, 0, stream>>>(
      actB, W3t, b3, nullptr, actA, Bq, Hq, Hq / 2);
  gemm_bf16_wmma<2><<<dim3(Bq / 128, Eq / 64), blk, 0, stream>>>(
      actA, W4t, b4, nullptr, zf, Bq, Hq / 2, Eq);                      // z fp32

  // ---- vector quantizer ----
  rownorm64_bf16<<<dim3(Bq / 8), blk, 0, stream>>>(zf, znb, Bq);
  vq_score_wmma<<<dim3(Bq / 128), blk, 0, stream>>>(znb, enb, idxb);
  vq_gather<<<dim3((Bq * Eq) / 256), blk, 0, stream>>>(idxb, emb, zf, qb, lossAcc, hist);

  // ---- decoder (q_ste == q in forward) ----
  gemm_bf16_wmma<0><<<dim3(Bq / 128, (Hq / 2) / 64), blk, 0, stream>>>(
      qb, D1t, db1, nullptr, actB, Bq, Eq, Hq / 2);
  gemm_bf16_wmma<0><<<dim3(Bq / 128, Hq / 64), blk, 0, stream>>>(
      actB, D2t, db2, nullptr, actA, Bq, Hq / 2, Hq);
  gemm_bf16_wmma<0><<<dim3(Bq / 128, Hq / 64), blk, 0, stream>>>(
      actA, D3t, db3, nullptr, actB, Bq, Hq, Hq);
  gemm_bf16_wmma<2><<<dim3(Bq / 128, Dq / 64), blk, 0, stream>>>(
      actB, D4t, db4, nullptr, outp + 1, Bq, Hq, Dq);                   // x_recon fp32

  // ---- scalars ----
  vq_finalize<<<dim3(1), blk, 0, stream>>>(lossAcc, hist, outp, outp + 1 + (size_t)Bq * Dq);
}
